// Quantizer_70875550319242
// MI455X (gfx1250) — compile-verified, matrix operand order checked by reading, not run on hardware
//
#include <hip/hip_runtime.h>
#include <hip/hip_bf16.h>

typedef __attribute__((ext_vector_type(16))) _Float16 v16h;
typedef __attribute__((ext_vector_type(8)))  _Float16 v8h;
typedef __attribute__((ext_vector_type(8)))  float    v8f;

// Assemble a 16x32-f16 WMMA A/B fragment for one lane from two 16-byte chunks.
__device__ inline v16h load_frag(const _Float16* p0, const _Float16* p1) {
    union { v16h v; v8h h[2]; } u;
    u.h[0] = *(const v8h*)p0;
    u.h[1] = *(const v8h*)p1;
    return u.v;
}

// --- Kernel 1: codebook prep: f16 copy + 0.5*||e||^2 --------------------------
__global__ __launch_bounds__(64) void vq_prep_emb(const float* __restrict__ emb,
                                                  _Float16* __restrict__ embh,
                                                  float* __restrict__ embn) {
    __shared__ float red[64];
    const int row = blockIdx.x;      // 1024 codes
    const int c   = threadIdx.x;     // 64 dims
    float e = emb[row * 64 + c];
    embh[row * 64 + c] = (_Float16)e;
    red[c] = e * e;
    __syncthreads();
    for (int s = 32; s > 0; s >>= 1) {
        if (c < s) red[c] += red[c + s];
        __syncthreads();
    }
    if (c == 0) embn[row] = 0.5f * red[0];
}

// --- Kernel 2: x [B,C,H,W] -> xt [N=B*H*W, 64] f16 row-major ------------------
__global__ __launch_bounds__(256) void vq_transpose(const float* __restrict__ x,
                                                    _Float16* __restrict__ xt) {
    __shared__ float tile[64][65];   // [c][w], padded
    const int b   = blockIdx.x >> 4;         // 64 batches
    const int hw0 = (blockIdx.x & 15) * 64;  // 16 tiles of 64 along H*W
    const int tid = threadIdx.x;
    for (int i = tid; i < 64 * 64; i += 256) {
        int c = i >> 6, w = i & 63;
        tile[c][w] = x[(size_t)b * 65536 + (size_t)c * 1024 + hw0 + w];
    }
    __syncthreads();
    for (int i = tid; i < 64 * 32; i += 256) {  // 2 halfs per thread (dword store)
        int w = i >> 5, cp = i & 31;
        union { _Float16 h[2]; unsigned u; } pk;
        pk.h[0] = (_Float16)tile[2 * cp + 0][w];
        pk.h[1] = (_Float16)tile[2 * cp + 1][w];
        *(unsigned*)&xt[((size_t)b * 1024 + hw0 + w) * 64 + 2 * cp] = pk.u;
    }
}

// --- Kernel 3: fused GEMM(argmax score) + gather + transposed write + loss ----
__global__ __launch_bounds__(256) void vq_main(const float* __restrict__ x,
                                               const float* __restrict__ emb,
                                               const _Float16* __restrict__ xt,
                                               const _Float16* __restrict__ embh,
                                               const float* __restrict__ embn,
                                               float* __restrict__ out_quant,
                                               float* __restrict__ out_idx,
                                               float* __restrict__ partials) {
    __shared__ int   idx_lds[128];
    __shared__ float red[256];

    const int tid  = threadIdx.x;
    const int wave = tid >> 5;
    const int lane = tid & 31;
    const int r    = lane & 15;       // row (A) / column (B/C/D) within 16
    const int hi   = lane >> 4;       // lane half selects K sub-chunks
    const int n0   = blockIdx.x * 128;          // 128 rows per block
    const int row  = n0 + wave * 16 + r;        // this lane's A row

    // A fragments for K = 0..31 and 32..63 (ISA 16-bit A layout: chunks at
    // K = 8*hi and 16+8*hi within each 32-wide fragment).
    const _Float16* xrow = xt + (size_t)row * 64;
    const v16h a0 = load_frag(xrow +      8 * hi, xrow + 16 + 8 * hi);
    const v16h a1 = load_frag(xrow + 32 + 8 * hi, xrow + 48 + 8 * hi);

    float bestv[8];
    int   besti[8];
#pragma unroll
    for (int v = 0; v < 8; ++v) { bestv[v] = -3.4e38f; besti[v] = 0; }

#pragma unroll 2
    for (int t = 0; t < 64; ++t) {              // 64 tiles of 16 codes
        const int col = t * 16 + r;
        const _Float16* erow = embh + (size_t)col * 64;  // B column n = emb row n
        const v16h b0 = load_frag(erow +      8 * hi, erow + 16 + 8 * hi);
        const v16h b1 = load_frag(erow + 32 + 8 * hi, erow + 48 + 8 * hi);

        // C/D layout: each lane holds one column -> seed accumulator with
        // -0.5*||e_col||^2 so the WMMA chain directly yields the argmax score.
        const float nb = -embn[col];
        v8f c;
#pragma unroll
        for (int v = 0; v < 8; ++v) c[v] = nb;
        c = __builtin_amdgcn_wmma_f32_16x16x32_f16(false, a0, false, b0,
                                                   (short)0, c, false, false);
        c = __builtin_amdgcn_wmma_f32_16x16x32_f16(false, a1, false, b1,
                                                   (short)0, c, false, false);
#pragma unroll
        for (int v = 0; v < 8; ++v) {
            if (c[v] > bestv[v]) { bestv[v] = c[v]; besti[v] = col; }
        }
    }

    // Max-reduce across the 16 lanes sharing each output row (wave32 shuffles;
    // xor masks 1,2,4,8 stay within each 16-lane half).
#pragma unroll
    for (int v = 0; v < 8; ++v) {
#pragma unroll
        for (int m = 1; m < 16; m <<= 1) {
            float ov = __shfl_xor(bestv[v], m, 32);
            int   oi = __shfl_xor(besti[v], m, 32);
            if (ov > bestv[v] || (ov == bestv[v] && oi < besti[v])) {
                bestv[v] = ov; besti[v] = oi;
            }
        }
        if (r == 0) idx_lds[wave * 16 + v + 8 * hi] = besti[v];  // M = v + 8*hi
    }
    __syncthreads();

    // Gather codebook rows, write quant in [B,C,H,W] layout (coalesced along hw),
    // accumulate MSE partial.
    const int b   = n0 >> 10;     // 128 | 1024, so a block stays in one batch
    const int hw0 = n0 & 1023;
    float lsum = 0.f;
    for (int i = tid; i < 64 * 128; i += 256) {
        int cc = i >> 7, j = i & 127;
        int k  = idx_lds[j];
        float q = emb[k * 64 + cc];
        size_t o = (size_t)b * 65536 + (size_t)cc * 1024 + hw0 + j;
        float d = q - x[o];
        lsum += d * d;
        out_quant[o] = q;
    }
    if (tid < 128) out_idx[n0 + tid] = (float)idx_lds[tid];

    red[tid] = lsum;
    __syncthreads();
    for (int s = 128; s > 0; s >>= 1) {
        if (tid < s) red[tid] += red[tid + s];
        __syncthreads();
    }
    if (tid == 0) partials[blockIdx.x] = red[0];
}

// --- Kernel 4: deterministic loss reduction -----------------------------------
__global__ __launch_bounds__(256) void vq_finalize(const float* __restrict__ partials,
                                                   float* __restrict__ out_loss) {
    __shared__ float red[256];
    const int tid = threadIdx.x;
    red[tid] = partials[tid] + partials[tid + 256];
    __syncthreads();
    for (int s = 128; s > 0; s >>= 1) {
        if (tid < s) red[tid] += red[tid + s];
        __syncthreads();
    }
    if (tid == 0) {
        float l = red[0] * (1.0f / 4194304.0f);   // mean over B*C*H*W
        out_loss[0] = l;   // codebook loss
        out_loss[1] = l;   // commitment loss (numerically identical)
    }
}

extern "C" void kernel_launch(void* const* d_in, const int* in_sizes, int n_in,
                              void* d_out, int out_size, void* d_ws, size_t ws_size,
                              hipStream_t stream) {
    (void)in_sizes; (void)n_in; (void)out_size; (void)ws_size;
    const float* x   = (const float*)d_in[0];   // [64,64,32,32]
    const float* emb = (const float*)d_in[1];   // [1024,64]

    char* ws = (char*)d_ws;
    _Float16* xt       = (_Float16*)ws;                               // 8 MB
    _Float16* embh     = (_Float16*)(ws + 8388608);                   // 128 KB
    float*    embn     = (float*)(ws + 8388608 + 131072);             // 4 KB
    float*    partials = (float*)(ws + 8388608 + 131072 + 4096);      // 2 KB

    float* out       = (float*)d_out;
    float* out_quant = out;                 // 4194304 floats [B,C,H,W]
    float* out_loss  = out + 4194304;       // 2 scalars
    float* out_idx   = out + 4194306;       // 65536 indices (as float)

    vq_prep_emb <<<1024,  64, 0, stream>>>(emb, embh, embn);
    vq_transpose<<<1024, 256, 0, stream>>>(x, xt);
    vq_main     <<< 512, 256, 0, stream>>>(x, emb, xt, embh, embn,
                                           out_quant, out_idx, partials);
    vq_finalize <<<   1, 256, 0, stream>>>(partials, out_loss);
}
